// SwinTransformerBlock_11948599018021
// MI455X (gfx1250) — matrix operand order repaired and checked by reading, not compiled, not run on hardware
//
#include <hip/hip_runtime.h>
#include <hip/hip_bf16.h>

typedef _Float16 v16h __attribute__((ext_vector_type(16)));
typedef _Float16 v8h  __attribute__((ext_vector_type(8)));
typedef float    v8f  __attribute__((ext_vector_type(8)));

#define Bn    32
#define Hn    56
#define Wn    56
#define Cn    384
#define NHn   12
#define WSn   14
#define SSn   7
#define Ln    (Hn*Wn)        /* 3136  */
#define Nn    (WSn*WSn)      /* 196   */
#define HDn   32
#define HIDn  1536
#define NWn   16
#define BWn   (Bn*NWn)       /* 512   */
#define MROWS (BWn*Nn)       /* 100352 */
#define SCALEQ 0.17677669529663687f  /* 32^-0.5 */

static const size_t QKV_STRIDE = (size_t)BWn * NHn * Nn * HDn; /* 38,535,168 elems */

// ---------------------------------------------------------------------------
// region id for the shifted-window attention mask (3x3 region grid)
__device__ __forceinline__ int region9(int sh, int sw) {
    int rh = (sh < Hn - WSn) ? 0 : ((sh < Hn - SSn) ? 1 : 2);
    int rw = (sw < Wn - WSn) ? 0 : ((sw < Wn - SSn) ? 1 : 2);
    return rh * 3 + rw;
}

// ---------------------------------------------------------------------------
// LayerNorm over C=384.  One wave per row; 8 rows per 256-thread block.
// mode==1: scatter output into shifted-window layout (for QKV input)
// mode==0: identity row layout (for FC1 input)
__global__ __launch_bounds__(256) void ln_kernel(
    const float* __restrict__ x, const float* __restrict__ w,
    const float* __restrict__ bb, _Float16* __restrict__ out, int mode)
{
    int wave = threadIdx.x >> 5;
    int lane = threadIdx.x & 31;
    int row  = blockIdx.x * 8 + wave;            // 0 .. 100351
    const float* xr = x + (size_t)row * Cn;

    float vals[12];
    float s = 0.f, s2 = 0.f;
#pragma unroll
    for (int i = 0; i < 12; ++i) {
        float t = xr[lane + i * 32];
        vals[i] = t; s += t; s2 += t * t;
    }
#pragma unroll
    for (int o = 16; o > 0; o >>= 1) {
        s  += __shfl_xor(s,  o, 32);
        s2 += __shfl_xor(s2, o, 32);
    }
    float mean = s * (1.f / Cn);
    float var  = s2 * (1.f / Cn) - mean * mean;
    float rstd = rsqrtf(var + 1e-5f);

    size_t dbase;
    if (mode) {
        int b  = row / Ln;
        int l  = row - b * Ln;
        int h0 = l / Wn, w0 = l - (l / Wn) * Wn;
        int sh = (h0 + Hn - SSn) % Hn;           // roll(-7)
        int sw = (w0 + Wn - SSn) % Wn;
        int wh = sh / WSn, qh = sh - wh * WSn;
        int ww = sw / WSn, qw = sw - ww * WSn;
        int b_ = b * NWn + wh * 4 + ww;
        int n  = qh * WSn + qw;
        dbase = ((size_t)b_ * Nn + n) * Cn;
    } else {
        dbase = (size_t)row * Cn;
    }
#pragma unroll
    for (int i = 0; i < 12; ++i) {
        int c = lane + i * 32;
        float y = (vals[i] - mean) * rstd * w[c] + bb[c];
        out[dbase + c] = (_Float16)y;
    }
}

// ---------------------------------------------------------------------------
// Generic WMMA GEMM: out[M,N] = A[M,K] @ W[N,K]^T  (f16 in, f32 accum)
// Block: 256 threads = 8 waves as 4(M) x 2(N); block tile 128x128,
// wave tile 32x64 (2 M-subtiles x 4 N-tiles); each B fragment feeds 2 WMMAs.
// EPI 0: QKV scatter (q*scale,k,v per-head) | 1: proj + reverse-window + resid
// EPI 2: FC1 + GELU -> f16                  | 3: FC2, d_out += v
template <int EPI>
__global__ __launch_bounds__(256) void wmma_gemm(
    const _Float16* __restrict__ A, const _Float16* __restrict__ W,
    const float* __restrict__ bias,
    _Float16* __restrict__ outh, float* __restrict__ outf,
    const float* __restrict__ resid, int M, int N, int K)
{
    int wave = threadIdx.x >> 5;
    int lane = threadIdx.x & 31;
    int wm   = wave & 3;                         // 0..3  (M)
    int wn   = wave >> 2;                        // 0..1  (N)
    int m0   = blockIdx.x * 128 + wm * 32;
    int n0   = blockIdx.y * 128 + wn * 64;
    int l15  = lane & 15;
    int kb8  = (lane >> 4) * 8;                  // unused; A frag also loads from kb16
    int kb16 = (lane >> 4) * 16;                 // B-frag K base (0 or 16)

    v8f acc[2][4];
#pragma unroll
    for (int mt = 0; mt < 2; ++mt)
#pragma unroll
        for (int t = 0; t < 4; ++t)
#pragma unroll
            for (int r = 0; r < 8; ++r) acc[mt][t][r] = 0.f;

    const _Float16* arow0 = A + (size_t)(m0 + l15) * K;
    const _Float16* arow1 = A + (size_t)(m0 + 16 + l15) * K;

    for (int kk = 0; kk < K; kk += 32) {
        __builtin_prefetch(arow0 + kk + 256, 0, 1);
        __builtin_prefetch(arow1 + kk + 256, 0, 1);
        v8h a0lo = *(const v8h*)(arow0 + kk + kb16);
        v8h a0hi = *(const v8h*)(arow0 + kk + kb16 + 8);
        v8h a1lo = *(const v8h*)(arow1 + kk + kb16);
        v8h a1hi = *(const v8h*)(arow1 + kk + kb16 + 8);
        v16h af0, af1;
#pragma unroll
        for (int i = 0; i < 8; ++i) {
            af0[i] = a0lo[i]; af0[8 + i] = a0hi[i];
            af1[i] = a1lo[i]; af1[8 + i] = a1hi[i];
        }
#pragma unroll
        for (int t = 0; t < 4; ++t) {
            int n = n0 + t * 16 + l15;
            v16h bf = *(const v16h*)(W + (size_t)n * K + kk + kb16);
            acc[0][t] = __builtin_amdgcn_wmma_f32_16x16x32_f16(
                false, af0, false, bf, (short)0, acc[0][t], false, false);
            acc[1][t] = __builtin_amdgcn_wmma_f32_16x16x32_f16(
                false, af1, false, bf, (short)0, acc[1][t], false, false);
        }
    }

    int rofs = (lane >> 4) * 8;                  // C/D: lanes 16-31 hold M=r+8
#pragma unroll
    for (int mt = 0; mt < 2; ++mt) {
#pragma unroll
        for (int t = 0; t < 4; ++t) {
            int c = n0 + t * 16 + l15;
            float bv = bias[c];
#pragma unroll
            for (int r = 0; r < 8; ++r) {
                int m = m0 + mt * 16 + r + rofs;
                float v = acc[mt][t][r] + bv;
                if (EPI == 0) {                  // QKV scatter
                    int three = c / Cn;
                    int jj    = c - three * Cn;
                    int hh    = jj >> 5, hd = jj & 31;
                    int b_    = m / Nn;
                    int n     = m - b_ * Nn;
                    if (three == 0) v *= SCALEQ;
                    size_t dst = (size_t)three * QKV_STRIDE +
                                 (((size_t)(b_ * NHn + hh)) * Nn + n) * HDn + hd;
                    outh[dst] = (_Float16)v;
                } else if (EPI == 1) {           // proj: reverse window + roll + resid
                    int b_  = m / Nn;
                    int n   = m - b_ * Nn;
                    int b   = b_ >> 4;
                    int win = b_ & 15;
                    int wh  = win >> 2, ww = win & 3;
                    int qh  = n / WSn, qw = n - (n / WSn) * WSn;
                    int sh  = wh * WSn + qh, sw = ww * WSn + qw;
                    int h0  = (sh + SSn) % Hn;
                    int w0  = (sw + SSn) % Wn;
                    size_t off = ((size_t)b * Ln + h0 * Wn + w0) * Cn + c;
                    outf[off] = resid[off] + v;
                } else if (EPI == 2) {           // FC1 + exact GELU
                    float g = 0.5f * v * (1.f + erff(v * 0.70710678118654752f));
                    outh[(size_t)m * N + c] = (_Float16)g;
                } else {                         // FC2: residual add into d_out
                    size_t off = (size_t)m * Cn + c;
                    outf[off] += v;
                }
            }
        }
    }
}

// ---------------------------------------------------------------------------
// Windowed attention: one wave per (16-query slab, head, window).
// S = q@k^T via WMMA -> LDS, fused rel-bias + shift-mask softmax * rscale
// (all 32 lanes: lane pair (l, l^16) splits each row), P (f16, LDS) @ V
// via WMMA -> per-head output f16 [B_,N,C] layout.
__global__ __launch_bounds__(32) void attn_kernel(
    const _Float16* __restrict__ qh, const _Float16* __restrict__ kh,
    const _Float16* __restrict__ vh, const float* __restrict__ rel_table,
    _Float16* __restrict__ oh)
{
    __shared__ float    S[16][224];
    __shared__ _Float16 P[16][224];

    int mt   = blockIdx.x;                       // 0..12  query tile
    int head = blockIdx.y;                       // 0..11
    int b_   = blockIdx.z;                       // 0..511
    int lane = threadIdx.x;
    int l15  = lane & 15;
    int half = lane >> 4;                        // 0 or 1
    int kb8  = half * 8;
    int kb16 = half * 16;
    int rofs = half * 8;

    size_t hb = ((size_t)(b_ * NHn + head)) * Nn * HDn;
    const _Float16* qb = qh + hb;
    const _Float16* kb = kh + hb;
    const _Float16* vb = vh + hb;

    // ---- A fragment: q slab 16x32 (clamped rows; pad rows zeroed later)
    int arow = mt * 16 + l15; if (arow > Nn - 1) arow = Nn - 1;
    v8h alo = *(const v8h*)(qb + arow * HDn + kb16);
    v8h ahi = *(const v8h*)(qb + arow * HDn + kb16 + 8);
    v16h af;
#pragma unroll
    for (int i = 0; i < 8; ++i) { af[i] = alo[i]; af[8 + i] = ahi[i]; }

    // ---- S = q @ k^T (13 N-tiles of 16 keys)
    for (int nt = 0; nt < 13; ++nt) {
        int kcol = nt * 16 + l15; if (kcol > Nn - 1) kcol = Nn - 1;
        v16h bf = *(const v16h*)(kb + kcol * HDn + kb16);  // 32B contiguous
        v8f cz;
#pragma unroll
        for (int r = 0; r < 8; ++r) cz[r] = 0.f;
        v8f sacc = __builtin_amdgcn_wmma_f32_16x16x32_f16(
            false, af, false, bf, (short)0, cz, false, false);
#pragma unroll
        for (int r = 0; r < 8; ++r)
            S[r + rofs][nt * 16 + l15] = sacc[r];
    }
    __syncthreads();

    // ---- bias + mask + softmax + rscale.  Row (l15) split across the lane
    // pair: half 0 handles keys [0,98), half 1 handles [98,196).
    {
        int q = mt * 16 + l15;
        if (q < Nn) {
            int qhh = q / WSn, qww = q - (q / WSn) * WSn;
            int win = b_ & 15;
            int wh  = win >> 2, ww = win & 3;
            int rq  = region9(wh * WSn + qhh, ww * WSn + qww);
            int k0  = half * 98, k1 = k0 + 98;
            float mx = -1e30f;
            for (int k_ = k0; k_ < k1; ++k_) {
                int khh = k_ / WSn, kww = k_ - (k_ / WSn) * WSn;
                int idx = (qhh - khh + WSn - 1) * (2 * WSn - 1) + (qww - kww + WSn - 1);
                float bias = rel_table[idx * (2 * NHn) + head];
                int rk = region9(wh * WSn + khh, ww * WSn + kww);
                float sv = S[l15][k_] + bias + ((rk == rq) ? 0.f : -100.f);
                S[l15][k_] = sv;
                mx = fmaxf(mx, sv);
            }
            mx = fmaxf(mx, __shfl_xor(mx, 16, 32));   // combine halves
            float sum = 0.f;
            for (int k_ = k0; k_ < k1; ++k_) {
                float e = __expf(S[l15][k_] - mx);
                S[l15][k_] = e;
                sum += e;
            }
            sum += __shfl_xor(sum, 16, 32);           // combine halves
            float inv = 1.f / sum;
            int p0 = half * 112, p1 = p0 + 112;       // cover padded width 224
            for (int k_ = p0; k_ < p1; ++k_) {
                if (k_ < Nn) {
                    int khh = k_ / WSn, kww = k_ - (k_ / WSn) * WSn;
                    int idx = (qhh - khh + WSn - 1) * (2 * WSn - 1) + (qww - kww + WSn - 1);
                    float rs = rel_table[idx * (2 * NHn) + NHn + head];
                    P[l15][k_] = (_Float16)(S[l15][k_] * inv * rs);
                } else {
                    P[l15][k_] = (_Float16)0.f;
                }
            }
        } else {
            int p0 = half * 112, p1 = p0 + 112;
            for (int k_ = p0; k_ < p1; ++k_) P[l15][k_] = (_Float16)0.f;
        }
    }
    __syncthreads();

    // ---- out = P @ V  (K padded to 224 -> 7 WMMA k-steps, 2 HD tiles)
    v8f oacc[2];
#pragma unroll
    for (int t = 0; t < 2; ++t)
#pragma unroll
        for (int r = 0; r < 8; ++r) oacc[t][r] = 0.f;

    for (int ks = 0; ks < 7; ++ks) {
        v8h plo = *(const v8h*)(&P[l15][ks * 32 + kb8]);
        v8h phi = *(const v8h*)(&P[l15][ks * 32 + kb8 + 16]);
        v16h pf;
#pragma unroll
        for (int i = 0; i < 8; ++i) { pf[i] = plo[i]; pf[8 + i] = phi[i]; }
#pragma unroll
        for (int vt = 0; vt < 2; ++vt) {
            int hd = vt * 16 + l15;
            v16h vf;
#pragma unroll
            for (int i = 0; i < 16; ++i) {
                int key = ks * 32 + kb8 + i + (i >> 3) * 8;
                if (key > Nn - 1) key = Nn - 1;     // safe (P is zero there)
                vf[i] = vb[key * HDn + hd];
            }
            oacc[vt] = __builtin_amdgcn_wmma_f32_16x16x32_f16(
                false, pf, false, vf, (short)0, oacc[vt], false, false);
        }
    }
#pragma unroll
    for (int vt = 0; vt < 2; ++vt) {
#pragma unroll
        for (int r = 0; r < 8; ++r) {
            int q = mt * 16 + r + rofs;
            if (q < Nn) {
                size_t off = ((size_t)b_ * Nn + q) * Cn + head * HDn + vt * 16 + l15;
                oh[off] = (_Float16)oacc[vt][r];
            }
        }
    }
}

// ---------------------------------------------------------------------------
extern "C" void kernel_launch(void* const* d_in, const int* in_sizes, int n_in,
                              void* d_out, int out_size, void* d_ws, size_t ws_size,
                              hipStream_t stream)
{
    const float* x   = (const float*)d_in[0];
    const float* n1w = (const float*)d_in[1];
    const float* n1b = (const float*)d_in[2];
    (void)in_sizes; (void)n_in; (void)out_size; (void)ws_size;

    // Workspace layout (bytes):
    //   [0]            xw / ln2-out f16 : 100352*384*2 = 77,070,336
    //   [77,070,336]   qkv f16 (3x)     : 231,211,008  (reused as FC1 out head)
    //   [308,281,344]  attn out f16     : 77,070,336   (tail of FC1 out reuse)
    //   [385,351,680]  f16 weight copies: (1152+384+1536)*384*2 + 384*1536*2
    char* ws = (char*)d_ws;
    _Float16* xw    = (_Float16*)(ws);
    _Float16* qkvb  = (_Float16*)(ws + 77070336ull);
    _Float16* ohb   = (_Float16*)(ws + 77070336ull + 231211008ull);
    _Float16* hbuf  = qkvb;                                 // FC1 output reuse
    char* wbase     = ws + 385351680ull;
    _Float16* qkvW  = (_Float16*)(wbase);                   // [1152,384]
    _Float16* projW = qkvW + (size_t)3 * Cn * Cn;           // [384,384]
    _Float16* fc1W  = projW + (size_t)Cn * Cn;              // [1536,384]
    _Float16* fc2W  = fc1W + (size_t)HIDn * Cn;             // [384,1536]
    float* out = (float*)d_out;

    // f32 -> f16 weight conversion
    struct Cvt {
        static __global__ void run(const float* s, _Float16* d, int n) {
            int i = blockIdx.x * blockDim.x + threadIdx.x;
            if (i < n) d[i] = (_Float16)s[i];
        }
    };
    int wq = 3 * Cn * Cn, wp = Cn * Cn, w1 = HIDn * Cn, w2 = Cn * HIDn;
    hipLaunchKernelGGL(Cvt::run, dim3((wq + 255) / 256), dim3(256), 0, stream,
                       (const float*)d_in[3], qkvW, wq);
    hipLaunchKernelGGL(Cvt::run, dim3((wp + 255) / 256), dim3(256), 0, stream,
                       (const float*)d_in[5], projW, wp);
    hipLaunchKernelGGL(Cvt::run, dim3((w1 + 255) / 256), dim3(256), 0, stream,
                       (const float*)d_in[10], fc1W, w1);
    hipLaunchKernelGGL(Cvt::run, dim3((w2 + 255) / 256), dim3(256), 0, stream,
                       (const float*)d_in[12], fc2W, w2);

    // 1) LN1 + shift + window partition -> xw (f16)
    ln_kernel<<<dim3(MROWS / 8), dim3(256), 0, stream>>>(x, n1w, n1b, xw, 1);

    // 2) QKV GEMM -> per-head q(*scale)/k/v
    wmma_gemm<0><<<dim3(MROWS / 128, (3 * Cn) / 128), dim3(256), 0, stream>>>(
        xw, qkvW, (const float*)d_in[4], qkvb, nullptr, nullptr,
        MROWS, 3 * Cn, Cn);

    // 3) windowed attention
    attn_kernel<<<dim3(13, NHn, BWn), dim3(32), 0, stream>>>(
        qkvb, qkvb + QKV_STRIDE, qkvb + 2 * QKV_STRIDE,
        (const float*)d_in[7], ohb);

    // 4) proj GEMM + reverse window/shift + shortcut -> d_out = x2 (f32)
    wmma_gemm<1><<<dim3(MROWS / 128, Cn / 128), dim3(256), 0, stream>>>(
        ohb, projW, (const float*)d_in[6], nullptr, out, x,
        MROWS, Cn, Cn);

    // 5) LN2 -> xw (f16, identity layout)
    ln_kernel<<<dim3(MROWS / 8), dim3(256), 0, stream>>>(
        out, (const float*)d_in[8], (const float*)d_in[9], xw, 0);

    // 6) FC1 + GELU -> hbuf (f16)
    wmma_gemm<2><<<dim3(MROWS / 128, HIDn / 128), dim3(256), 0, stream>>>(
        xw, fc1W, (const float*)d_in[11], hbuf, nullptr, nullptr,
        MROWS, HIDn, Cn);

    // 7) FC2, d_out += h
    wmma_gemm<3><<<dim3(MROWS / 128, Cn / 128), dim3(256), 0, stream>>>(
        hbuf, fc2W, (const float*)d_in[13], nullptr, out, nullptr,
        MROWS, Cn, HIDn);
}
